// LocalDirectionalAttention_89421219103598
// MI455X (gfx1250) — compile-verified, matrix operand order checked by reading, not run on hardware
//
#include <hip/hip_runtime.h>

#define NEGV (-1.0e9f)
#define LDA 72   // padded LDS row stride for bf16 64-col tiles (144B: conflict-free, 16B-aligned)
#define LDS_SC 65 // padded row stride for f32 64-col scores (260B: conflict-free)

typedef __bf16 bf16;
typedef __attribute__((ext_vector_type(8)))  __bf16 v8bf;
typedef __attribute__((ext_vector_type(16))) __bf16 v16bf;
typedef __attribute__((ext_vector_type(8)))  float  v8f;

static __device__ __forceinline__ v8f zero8() {
  v8f z;
#pragma unroll
  for (int i = 0; i < 8; ++i) z[i] = 0.0f;
  return z;
}

// Build one 16-element bf16 WMMA fragment for this lane from a contiguous row.
// Per the CDNA5 16-bit A/B layout, each lane needs K = {k0..k0+7} and
// {k0+16..k0+23}: two contiguous 16-byte loads.
static __device__ __forceinline__ v16bf load_frag(const bf16* p) {
  v8bf lo = *reinterpret_cast<const v8bf*>(p);
  v8bf hi = *reinterpret_cast<const v8bf*>(p + 16);
  v16bf r;
#pragma unroll
  for (int i = 0; i < 8; ++i) { r[i] = lo[i]; r[i + 8] = hi[i]; }
  return r;
}

// ---------------------------------------------------------------------------
// LDS-staged GEMM core: block tile M=64 x N=(8 waves * 32), K=768.
// A panel (shared by all 8 waves) staged 64x64 per chunk into LDS with
// double buffering; waves read A fragments via ds_load, B from global.
// ---------------------------------------------------------------------------
static __device__ __forceinline__ void gemm_lds_core(
    const bf16* __restrict__ Ag, const bf16* __restrict__ B0,
    const bf16* __restrict__ B1, bf16* sA /* [2][64*LDA] */,
    int tid, v8f (&acc)[4][2]) {
  int lane = tid & 31;
  int r  = lane & 15;
  int kh = (lane >> 4) << 3;                // 0 or 8

  // cooperative staging: 256 threads x 16 elements = 64x64 chunk
  int lrow = tid >> 2;                      // 0..63
  int lcol = (tid & 3) << 4;                // 0,16,32,48
  const bf16* gsrc = Ag + (size_t)lrow * 768 + lcol;
  bf16* ldst = sA + lrow * LDA + lcol;

  auto stage = [&](int buf, int kb) {
    v8bf d0 = *reinterpret_cast<const v8bf*>(gsrc + kb);
    v8bf d1 = *reinterpret_cast<const v8bf*>(gsrc + kb + 8);
    *reinterpret_cast<v8bf*>(ldst + buf * 64 * LDA)     = d0;
    *reinterpret_cast<v8bf*>(ldst + buf * 64 * LDA + 8) = d1;
  };

  stage(0, 0);
  __syncthreads();
  int buf = 0;
  for (int kb = 0; kb < 768; kb += 64) {
    if (kb + 64 < 768) stage(buf ^ 1, kb + 64);  // prefetch next chunk
    const bf16* sAb = sA + buf * 64 * LDA;
#pragma unroll
    for (int kc = 0; kc < 2; ++kc) {             // two K=32 sub-chunks
      v16bf b0 = load_frag(B0 + kb + kc * 32 + kh);
      v16bf b1 = load_frag(B1 + kb + kc * 32 + kh);
#pragma unroll
      for (int mt = 0; mt < 4; ++mt) {
        v16bf a = load_frag(sAb + (mt * 16 + r) * LDA + kc * 32 + kh);
        acc[mt][0] = __builtin_amdgcn_wmma_f32_16x16x32_bf16(
            false, a, false, b0, (short)0, acc[mt][0], false, false);
        acc[mt][1] = __builtin_amdgcn_wmma_f32_16x16x32_bf16(
            false, a, false, b1, (short)0, acc[mt][1], false, false);
      }
    }
    __syncthreads();                             // staging buf^1 done + reads of buf done
    buf ^= 1;
  }
}

// ---------------------------------------------------------------------------
// Stage 1a: gather patches into window-major padded bf16 [512][64][768]
// ---------------------------------------------------------------------------
__global__ __launch_bounds__(256) void pack_windows(const float* __restrict__ x,
                                                    bf16* __restrict__ win) {
  int w = blockIdx.x;                       // 0..511 : b*4 + wh*2 + ww
  int b = w >> 2, wh = (w >> 1) & 1, ww = w & 1;
  const float* xb = x + (size_t)b * 197 * 768;
  bf16* dst = win + (size_t)w * 64 * 768;
  for (int idx = threadIdx.x; idx < 64 * 768; idx += 256) {
    int row = idx / 768, ch = idx - row * 768;
    float v = 0.0f;
    if (row < 49) {
      int i = row / 7, j = row - i * 7;
      int t = 1 + (wh * 7 + i) * 14 + (ww * 7 + j);
      v = xb[(size_t)t * 768 + ch];
    }
    dst[idx] = (bf16)v;
  }
}

// Stage 1b: pass cls tokens straight through to the output
__global__ __launch_bounds__(256) void copy_cls(const float* __restrict__ x,
                                                float* __restrict__ out) {
  int idx = blockIdx.x * 256 + threadIdx.x;
  if (idx < 128 * 768) {
    int b = idx / 768, ch = idx - b * 768;
    out[(size_t)b * 197 * 768 + ch] = x[(size_t)b * 197 * 768 + ch];
  }
}

// Stage 1c: weights -> bf16. wcat = [Wq ; Wk(5) ; Wv] flat (rows are the
// B^T rows the WMMA B-fragment wants). bcat = [bq ; bk(5) ; bv] f32.
__global__ __launch_bounds__(256) void convert_weights(
    const float* __restrict__ Wq, const float* __restrict__ Wk,
    const float* __restrict__ Wv, const float* __restrict__ Wp,
    const float* __restrict__ bq, const float* __restrict__ bk,
    const float* __restrict__ bv,
    bf16* __restrict__ wcat, bf16* __restrict__ wpc, float* __restrict__ bcat) {
  const int NE = 768 * 768;
  int stride = gridDim.x * blockDim.x;
  int i0 = blockIdx.x * blockDim.x + threadIdx.x;
  for (int i = i0; i < 7 * NE; i += stride) {
    float v = (i < NE) ? Wq[i] : (i < 6 * NE ? Wk[i - NE] : Wv[i - 6 * NE]);
    wcat[i] = (bf16)v;
  }
  for (int i = i0; i < NE; i += stride) wpc[i] = (bf16)Wp[i];
  if (i0 < 5376)
    bcat[i0] = (i0 < 768) ? bq[i0] : (i0 < 4608 ? bk[i0 - 768] : bv[i0 - 4608]);
}

// ---------------------------------------------------------------------------
// Stage 2: fused QKV projection GEMM  (M=64 per window, K=768, N=5376)
// grid (512, 21), 256 threads = 8 waves; wave owns a 64x32 tile.
// ---------------------------------------------------------------------------
__global__ __launch_bounds__(256, 4)
void qkv_gemm(const bf16* __restrict__ win,
    const bf16* __restrict__ wcat, const float* __restrict__ bcat,
    bf16* __restrict__ qb, bf16* __restrict__ kb, bf16* __restrict__ vt) {
  __shared__ __align__(16) bf16 sA[2 * 64 * LDA];
  int w = blockIdx.x;
  int tid = threadIdx.x;
  int lane = tid & 31, wave = tid >> 5;
  int n0 = blockIdx.y * 256 + wave * 32;
  int r  = lane & 15;

  const bf16* A  = win + (size_t)w * 64 * 768;
  const bf16* B0 = wcat + (size_t)(n0 + r) * 768;
  const bf16* B1 = wcat + (size_t)(n0 + 16 + r) * 768;

  v8f acc[4][2];
#pragma unroll
  for (int mt = 0; mt < 4; ++mt) { acc[mt][0] = zero8(); acc[mt][1] = zero8(); }

  gemm_lds_core(A, B0, B1, sA, tid, acc);

  int rowadd = (lane >> 4) << 3;
#pragma unroll
  for (int s = 0; s < 2; ++s) {
    int colg = n0 + s * 16 + r;
    float bias = bcat[colg];
#pragma unroll
    for (int mt = 0; mt < 4; ++mt)
#pragma unroll
      for (int j = 0; j < 8; ++j) {
        int row = mt * 16 + j + rowadd;
        float v = acc[mt][s][j] + bias;
        if (colg < 768) {                      // Q, pre-scaled by 1/sqrt(d)
          qb[((size_t)w * 64 + row) * 768 + colg] = (bf16)(v * 0.125f);
        } else if (colg < 4608) {              // K for 5 directions
          int cc = colg - 768;
          int dir = cc / 768, ch = cc - dir * 768;
          kb[(((size_t)dir * 512 + w) * 64 + row) * 768 + ch] = (bf16)v;
        } else {                               // V, stored transposed [h][d][m]
          int ch = colg - 4608;
          int hh = ch >> 6, dcol = ch & 63;
          vt[(((size_t)w * 12 + hh) * 64 + dcol) * 64 + row] = (bf16)v;
        }
      }
  }
}

// ---------------------------------------------------------------------------
// Stage 3: masked 5-direction attention, one wave per (window, head)
// ---------------------------------------------------------------------------
__global__ __launch_bounds__(32) void attention(const bf16* __restrict__ qb,
    const bf16* __restrict__ kbuf, const bf16* __restrict__ vt,
    bf16* __restrict__ ab) {
  __shared__ __align__(16) float s_sc[64 * LDS_SC]; // scores f32 (padded stride)
  __shared__ __align__(16) bf16  s_at[64 * LDA];    // softmaxed attn bf16 (padded)
  int w = blockIdx.x, h = blockIdx.y;
  int lane = threadIdx.x;
  int r = lane & 15;
  int kh = (lane >> 4) << 3;
  int rowadd = kh;

  const bf16* Q = qb + (size_t)w * 64 * 768 + h * 64;
  v16bf aq[4][2];
#pragma unroll
  for (int mt = 0; mt < 4; ++mt)
#pragma unroll
    for (int kc = 0; kc < 2; ++kc)
      aq[mt][kc] = load_frag(Q + (size_t)(mt * 16 + r) * 768 + kc * 32 + kh);

  // V fragments are dir-invariant: load once.
  const bf16* V = vt + (size_t)(w * 12 + h) * 4096;
  v16bf bv_[4][2];
#pragma unroll
  for (int nt = 0; nt < 4; ++nt)
#pragma unroll
    for (int kc = 0; kc < 2; ++kc)
      bv_[nt][kc] = load_frag(V + (nt * 16 + r) * 64 + kc * 32 + kh);

  v8f outp[16];
#pragma unroll
  for (int t = 0; t < 16; ++t) outp[t] = zero8();

  for (int dir = 0; dir < 5; ++dir) {
    const bf16* K = kbuf + (((size_t)dir * 512 + w) * 64) * 768 + h * 64;
    v16bf bk[4][2];
#pragma unroll
    for (int nt = 0; nt < 4; ++nt)
#pragma unroll
      for (int kc = 0; kc < 2; ++kc)
        bk[nt][kc] = load_frag(K + (size_t)(nt * 16 + r) * 768 + kc * 32 + kh);

    // S = Q K^T  (64x64 padded), add directional mask, dump to LDS
#pragma unroll
    for (int mt = 0; mt < 4; ++mt)
#pragma unroll
      for (int nt = 0; nt < 4; ++nt) {
        v8f s = __builtin_amdgcn_wmma_f32_16x16x32_bf16(
            false, aq[mt][0], false, bk[nt][0], (short)0, zero8(), false, false);
        s = __builtin_amdgcn_wmma_f32_16x16x32_bf16(
            false, aq[mt][1], false, bk[nt][1], (short)0, s, false, false);
#pragma unroll
        for (int j = 0; j < 8; ++j) {
          int rr = mt * 16 + j + rowadd;     // query index
          int cc = nt * 16 + r;              // key index
          bool ok = false;
          if (rr < 49 && cc < 49) {
            int qr = rr / 7, qc = rr - qr * 7;
            int kr = cc / 7, kc2 = cc - kr * 7;
            ok = (dir == 0) ? (kc2 < qc)
               : (dir == 1) ? (kc2 > qc)
               : (dir == 2) ? (kr  < qr)
               : (dir == 3) ? (kr  > qr)
                            : (rr == cc);
          }
          s_sc[rr * LDS_SC + cc] = ok ? s[j] : (s[j] + NEGV);
        }
      }
    __syncthreads();

    // row-wise softmax (2 rows per lane)
#pragma unroll
    for (int rep = 0; rep < 2; ++rep) {
      int row = lane + rep * 32;
      float* sp = s_sc + row * LDS_SC;
      float mx = -3.4e38f;
      for (int c = 0; c < 64; ++c) mx = fmaxf(mx, sp[c]);
      float sum = 0.0f;
      for (int c = 0; c < 64; ++c) { float e = __expf(sp[c] - mx); sp[c] = e; sum += e; }
      float inv = 1.0f / sum;
      bf16* ap = s_at + row * LDA;
      for (int c = 0; c < 64; ++c) ap[c] = (bf16)(sp[c] * inv);
    }
    __syncthreads();

    // out += attn @ V   (V pre-transposed per head: Vt[d][m])
#pragma unroll
    for (int mt = 0; mt < 4; ++mt) {
      v16bf a0 = load_frag(&s_at[(mt * 16 + r) * LDA + kh]);
      v16bf a1 = load_frag(&s_at[(mt * 16 + r) * LDA + 32 + kh]);
#pragma unroll
      for (int nt = 0; nt < 4; ++nt) {
        outp[mt * 4 + nt] = __builtin_amdgcn_wmma_f32_16x16x32_bf16(
            false, a0, false, bv_[nt][0], (short)0, outp[mt * 4 + nt], false, false);
        outp[mt * 4 + nt] = __builtin_amdgcn_wmma_f32_16x16x32_bf16(
            false, a1, false, bv_[nt][1], (short)0, outp[mt * 4 + nt], false, false);
      }
    }
    __syncthreads();
  }

  // write (sum over 5 dirs) / 5 as bf16 for the final projection GEMM
#pragma unroll
  for (int mt = 0; mt < 4; ++mt)
#pragma unroll
    for (int nt = 0; nt < 4; ++nt)
#pragma unroll
      for (int j = 0; j < 8; ++j) {
        int rr = mt * 16 + j + rowadd;
        int cc = nt * 16 + r;
        ab[((size_t)w * 64 + rr) * 768 + h * 64 + cc] =
            (bf16)(outp[mt * 4 + nt][j] * 0.2f);
      }
}

// ---------------------------------------------------------------------------
// Stage 4: output projection GEMM + un-window scatter into (B,197,C) f32
// grid (512, 3), 256 threads = 8 waves; wave owns a 64x32 tile.
// ---------------------------------------------------------------------------
__global__ __launch_bounds__(256, 4)
void proj_gemm(const bf16* __restrict__ ab,
    const bf16* __restrict__ wpc, const float* __restrict__ bp,
    float* __restrict__ out) {
  __shared__ __align__(16) bf16 sA[2 * 64 * LDA];
  int w = blockIdx.x;
  int tid = threadIdx.x;
  int lane = tid & 31, wave = tid >> 5;
  int n0 = blockIdx.y * 256 + wave * 32;
  int r = lane & 15;

  const bf16* A  = ab + (size_t)w * 64 * 768;
  const bf16* B0 = wpc + (size_t)(n0 + r) * 768;
  const bf16* B1 = wpc + (size_t)(n0 + 16 + r) * 768;

  v8f acc[4][2];
#pragma unroll
  for (int mt = 0; mt < 4; ++mt) { acc[mt][0] = zero8(); acc[mt][1] = zero8(); }

  gemm_lds_core(A, B0, B1, sA, tid, acc);

  int b = w >> 2, wh = (w >> 1) & 1, ww = w & 1;
  int rowadd = (lane >> 4) << 3;
#pragma unroll
  for (int s = 0; s < 2; ++s) {
    int colg = n0 + s * 16 + r;
    float bias = bp[colg];
#pragma unroll
    for (int mt = 0; mt < 4; ++mt)
#pragma unroll
      for (int j = 0; j < 8; ++j) {
        int row = mt * 16 + j + rowadd;
        if (row < 49) {
          int i = row / 7, jj = row - i * 7;
          int t = 1 + (wh * 7 + i) * 14 + (ww * 7 + jj);
          out[((size_t)b * 197 + t) * 768 + colg] = acc[mt][s][j] + bias;
        }
      }
  }
}

// ---------------------------------------------------------------------------
extern "C" void kernel_launch(void* const* d_in, const int* in_sizes, int n_in,
                              void* d_out, int out_size, void* d_ws, size_t ws_size,
                              hipStream_t stream) {
  (void)in_sizes; (void)n_in; (void)out_size; (void)ws_size;
  const float* x  = (const float*)d_in[0];
  const float* Wq = (const float*)d_in[1];
  const float* bq = (const float*)d_in[2];
  const float* Wk = (const float*)d_in[3];
  const float* bk = (const float*)d_in[4];
  const float* Wv = (const float*)d_in[5];
  const float* bv = (const float*)d_in[6];
  const float* Wp = (const float*)d_in[7];
  const float* bp = (const float*)d_in[8];
  float* out = (float*)d_out;

  char* ws = (char*)d_ws;
  size_t off = 0;
  auto alloc = [&](size_t bytes) {
    char* p = ws + off;
    off += (bytes + 255) & ~(size_t)255;
    return p;
  };
  bf16*  win  = (bf16*) alloc((size_t)512 * 64 * 768 * 2);        // packed windows
  bf16*  wcat = (bf16*) alloc((size_t)5376 * 768 * 2);            // Wq|Wk*5|Wv bf16
  bf16*  wpc  = (bf16*) alloc((size_t)768 * 768 * 2);             // Wp bf16
  float* bcat = (float*)alloc((size_t)5376 * 4);                  // bq|bk*5|bv
  bf16*  qb   = (bf16*) alloc((size_t)512 * 64 * 768 * 2);        // Q (scaled)
  bf16*  kb   = (bf16*) alloc((size_t)5 * 512 * 64 * 768 * 2);    // K x 5 dirs
  bf16*  vt   = (bf16*) alloc((size_t)512 * 12 * 64 * 64 * 2);    // V^T per head
  bf16*  ab   = (bf16*) alloc((size_t)512 * 64 * 768 * 2);        // attn out

  pack_windows<<<512, 256, 0, stream>>>(x, win);
  copy_cls<<<(128 * 768 + 255) / 256, 256, 0, stream>>>(x, out);
  convert_weights<<<1024, 256, 0, stream>>>(Wq, Wk, Wv, Wp, bq, bk, bv,
                                            wcat, wpc, bcat);
  qkv_gemm<<<dim3(512, 21), 256, 0, stream>>>(win, wcat, bcat, qb, kb, vt);
  attention<<<dim3(512, 12), 32, 0, stream>>>(qb, kb, vt, ab);
  proj_gemm<<<dim3(512, 3), 256, 0, stream>>>(ab, wpc, bp, out);
}
